// MambaCrossModel_360777252923
// MI455X (gfx1250) — matrix operand order repaired and checked by reading, not compiled, not run on hardware
//
#include <hip/hip_runtime.h>
#include <hip/hip_bf16.h>
#include <stdint.h>

// ---- problem dims (match reference) ----
#define B_SZ  4
#define L_SZ  1024
#define DM    1024           // d_model
#define DI    2048           // d_inner
#define NST   16             // d_state
#define DCONV 4
#define DTR   64             // dt_rank
#define NTOK  (B_SZ * L_SZ)  // 4096 tokens
#define DBCW  (DTR + 2 * NST) // 96

typedef __attribute__((ext_vector_type(16))) __bf16 v16bf;
typedef __attribute__((ext_vector_type(8)))  __bf16 v8bf;
typedef __attribute__((ext_vector_type(8)))  float  v8f;
typedef unsigned int u32x4 __attribute__((ext_vector_type(4)));
typedef unsigned int u32x8 __attribute__((ext_vector_type(8)));

// ---------------------------------------------------------------------------
// TDM: one-instruction DMA of a 2D tile (rows x elems of 2-byte data) into
// LDS, with hardware row padding 64B->80B to match the LDS tile stride.
// Descriptor per CDNA5 ISA (D# groups 0 and 1, 2D form,
// VADDR2/VADDR3 = NULL). EXEC is ignored; issue from a single wave.
// ---------------------------------------------------------------------------
__device__ __attribute__((always_inline)) inline void tdm_load_tile_2d(
    unsigned lds_addr, const void* gaddr,
    unsigned avail0_elems,   // tensor_dim0: readable elems per row (OOB -> 0)
    unsigned avail1_rows,    // tensor_dim1: readable rows        (OOB -> 0)
    unsigned tile0_elems,    // tile_dim0 (elems per row; 64B here)
    unsigned tile1_rows,     // tile_dim1
    unsigned stride0_elems)  // row stride in elems
{
  uint64_t ga = (uint64_t)(uintptr_t)gaddr;
  u32x4 g0;
  g0[0] = 1u;                                        // count=1: valid user D#
  g0[1] = lds_addr;                                  // LDS byte address
  g0[2] = (unsigned)ga;                              // global_addr[31:0]
  g0[3] = (unsigned)((ga >> 32) & 0x01FFFFFFu)       // global_addr[56:32]
        | (2u << 30);                                // type = 2 ("image")
  u32x8 g1;
  g1[0] = (1u << 16)      // data_size = 1 -> 2 bytes
        | (1u << 20)      // pad_enable
        | (3u << 22)      // pad_interval = 3 -> every 16 DWORDs (64B)
        | (3u << 25);     // pad_amount   = 3 -> 4 DWORDs (16B)  => 80B stride
  g1[1] = (avail0_elems & 0xFFFFu) << 16;                       // dim0[15:0]
  g1[2] = (avail0_elems >> 16) | ((avail1_rows & 0xFFFFu) << 16);
  g1[3] = (avail1_rows >> 16) | (tile0_elems << 16);            // tile_dim0
  g1[4] = tile1_rows;                                // tile_dim1; tile_dim2=0
  g1[5] = stride0_elems;                             // dim0_stride[31:0]
  g1[6] = 0u;                                        // stride hi / dim1_stride
  g1[7] = 0u;
  asm volatile("tensor_load_to_lds %0, %1" :: "s"(g0), "s"(g1) : "memory");
}

// ---------------------------------------------------------------------------
// RMSNorm -> bf16 (one block per token row)
// ---------------------------------------------------------------------------
__global__ __launch_bounds__(256) void k_rmsnorm_bf16(
    const float* __restrict__ x, const float* __restrict__ w,
    __bf16* __restrict__ out, int D)
{
  const int row = blockIdx.x;
  const float* xr = x + (size_t)row * D;
  float ss = 0.f;
  for (int i = threadIdx.x; i < D; i += 256) { float v = xr[i]; ss += v * v; }
  #pragma unroll
  for (int o = 16; o > 0; o >>= 1) ss += __shfl_down(ss, o, 32);
  __shared__ float wsum[8];
  __shared__ float s_scale;
  if ((threadIdx.x & 31) == 0) wsum[threadIdx.x >> 5] = ss;
  __syncthreads();
  if (threadIdx.x == 0) {
    float t = 0.f;
    #pragma unroll
    for (int i = 0; i < 8; ++i) t += wsum[i];
    s_scale = rsqrtf(t / (float)D + 1e-5f);
  }
  __syncthreads();
  const float sc = s_scale;
  __bf16* orow = out + (size_t)row * D;
  for (int i = threadIdx.x; i < D; i += 256)
    orow[i] = (__bf16)(xr[i] * sc * w[i]);
}

// ---------------------------------------------------------------------------
// strided f32 -> bf16 convert (weights, d_r slice)
// ---------------------------------------------------------------------------
__global__ __launch_bounds__(256) void k_f32_to_bf16(
    const float* __restrict__ src, int srcPitch,
    __bf16* __restrict__ dst, int dstPitch, int rows, int cols)
{
  size_t idx = (size_t)blockIdx.x * 256 + threadIdx.x;
  size_t total = (size_t)rows * cols;
  if (idx >= total) return;
  size_t r = idx / cols, c = idx % cols;
  dst[r * (size_t)dstPitch + c] = (__bf16)src[r * (size_t)srcPitch + c];
}

// ---------------------------------------------------------------------------
// bf16 WMMA GEMM:  C[M,N] = A[M,K] * W[N,K]^T  (f32 accumulate)
// block tile 128x64, K-step 32, 8 waves (4 along M x 2 along N),
// each wave: 32x32 output via 2x2 v_wmma_f32_16x16x32_bf16.
// Double-buffered tiles: A tile via TDM tensor_load_to_lds (TENSORcnt),
// B tile via per-lane GLOBAL_LOAD_ASYNC_TO_LDS_B128 (ASYNCcnt).
// ---------------------------------------------------------------------------
#define TM 128
#define TN 64
#define TK 32

__global__ __launch_bounds__(256) void k_gemm_bf16_wmma(
    const __bf16* __restrict__ A,   // M x K row-major
    const __bf16* __restrict__ W,   // N x K row-major
    float* __restrict__ C,          // M x N row-major
    int M, int N, int K)
{
  // +8 pad keeps row stride = 80 bytes (5 x 16B): 128-bit LDS ops stay
  // aligned and the TDM pad fields reproduce the same stride.
  __shared__ __bf16 sA[2][TM][TK + 8];
  __shared__ __bf16 sB[2][TN][TK + 8];

  const int tid   = threadIdx.x;
  const int lane  = tid & 31;
  const int wave  = tid >> 5;
  const int waveM = wave & 3;     // 4 waves along M
  const int waveN = wave >> 2;    // 2 waves along N
  const int m0 = waveM * 32;
  const int n0 = waveN * 32;
  const int blockM = blockIdx.y * TM;
  const int blockN = blockIdx.x * TN;
  const int g  = lane >> 4;       // lane half
  const int mr = lane & 15;
  const int ac = (tid & 3) * 8;   // K-chunk column (16B granules)

  v8f acc[2][2] = {};

  // issue tile fills for K-offset k0 into buffer `buf`
  auto load_tiles = [&](int k0, int buf) {
    // A tile (128x32 bf16): one TDM descriptor, issued by thread 0's wave.
    if (tid == 0) {
      tdm_load_tile_2d((unsigned)(uintptr_t)&sA[buf][0][0],
                       A + (size_t)blockM * K + k0,
                       (unsigned)(K - k0), (unsigned)(M - blockM),
                       TK, TM, (unsigned)K);
    }
    // B tile (64x32 bf16): per-lane async scatter; zero-fill ragged rows.
    {
      int r  = tid >> 2;
      int gr = blockN + r;
      unsigned lds = (unsigned)(uintptr_t)&sB[buf][r][ac];
      if (gr < N) {
        uint64_t gaddr = (uint64_t)(uintptr_t)(W + (size_t)gr * K + k0 + ac);
        asm volatile("global_load_async_to_lds_b128 %0, %1, off"
                     :: "v"(lds), "v"(gaddr) : "memory");
      } else {
        *(v8bf*)&sB[buf][r][ac] = (v8bf){};
      }
    }
  };

  // prologue: fill buffer 0
  load_tiles(0, 0);
  asm volatile("s_wait_asynccnt 0x0" ::: "memory");
  if (tid == 0) __builtin_amdgcn_s_wait_tensorcnt(0);
  __syncthreads();

  int cur = 0;
  for (int k0 = 0; k0 < K; k0 += TK) {
    // stream tile k0+TK into the other buffer while we compute this one
    if (k0 + TK < K) {
      load_tiles(k0 + TK, cur ^ 1);
      // L2 prefetch two tiles ahead -> global_prefetch_b8
      if (k0 + 2 * TK < K) {
        int gr = blockM + (tid >> 1);
        if (gr < M)
          __builtin_prefetch(A + (size_t)gr * K + k0 + 2 * TK + (tid & 1) * 16, 0, 1);
      }
    }

    // A fragment, contiguous k order shared with B:
    //   K = g*16 + e
    // B fragment, same k order as A: K = g*16 + e, N = mr
    v16bf af[2], bfv[2];
    #pragma unroll
    for (int t = 0; t < 2; ++t) {
      #pragma unroll
      for (int e = 0; e < 16; ++e) {
        int ka = g * 16 + e;
        af[t][e] = sA[cur][m0 + t * 16 + mr][ka];
      }
      #pragma unroll
      for (int e = 0; e < 16; ++e) {
        bfv[t][e] = sB[cur][n0 + t * 16 + mr][g * 16 + e];
      }
    }
    #pragma unroll
    for (int ti = 0; ti < 2; ++ti)
      #pragma unroll
      for (int tj = 0; tj < 2; ++tj)
        acc[ti][tj] = __builtin_amdgcn_wmma_f32_16x16x32_bf16(
            false, af[ti], false, bfv[tj], (short)0, acc[ti][tj], false, false);

    // drain my fills, then wait for everyone to stop reading `cur`
    asm volatile("s_wait_asynccnt 0x0" ::: "memory");
    if (tid == 0) __builtin_amdgcn_s_wait_tensorcnt(0);
    __syncthreads();
    cur ^= 1;
  }

  // C/D layout: VGPR r -> M = g*8 + r, N = lane&15
  #pragma unroll
  for (int ti = 0; ti < 2; ++ti)
    #pragma unroll
    for (int tj = 0; tj < 2; ++tj)
      #pragma unroll
      for (int r = 0; r < 8; ++r) {
        int row = blockM + m0 + ti * 16 + g * 8 + r;
        int col = blockN + n0 + tj * 16 + mr;
        if (row < M && col < N) C[(size_t)row * N + col] = acc[ti][tj][r];
      }
}

// ---------------------------------------------------------------------------
// depthwise causal conv over L (+bias); optional bf16 mirror for next GEMM
// ---------------------------------------------------------------------------
__global__ __launch_bounds__(256) void k_dwconv_causal(
    const float* __restrict__ x, int xPitch,
    const float* __restrict__ w, const float* __restrict__ bias,
    float* __restrict__ y, __bf16* __restrict__ ybf)
{
  size_t idx = (size_t)blockIdx.x * 256 + threadIdx.x;
  const size_t total = (size_t)NTOK * DI;
  if (idx >= total) return;
  int d = (int)(idx % DI);
  size_t bl = idx / DI;
  int l = (int)(bl % L_SZ);
  size_t b = bl / L_SZ;
  float acc = bias[d];
  const float* wd = w + (size_t)d * DCONV;
  #pragma unroll
  for (int j = 0; j < DCONV; ++j) {
    int ls = l - (DCONV - 1) + j;
    if (ls >= 0) acc += wd[j] * x[(b * L_SZ + ls) * (size_t)xPitch + d];
  }
  y[idx] = acc;
  if (ybf) ybf[idx] = (__bf16)acc;
}

// ---------------------------------------------------------------------------
// bidirectional selective scan: one thread per (b, d) channel.
// fwd uses A2, bwd uses A2b with u reversed; delta/B/C not reversed;
// bwd output written reversed back. delta = softplus(delta_raw + b_dt).
// ---------------------------------------------------------------------------
__global__ __launch_bounds__(256) void k_selective_scan_bidir(
    const float* __restrict__ u,          // (B,L,DI)
    const float* __restrict__ delta_raw,  // (B,L,DI)
    const float* __restrict__ dbc,        // (B,L,96): [64,80)=B, [80,96)=C
    const float* __restrict__ A_log,      // (DI,16)
    const float* __restrict__ Ab_log,     // (DI,16)
    const float* __restrict__ b_dt,       // (DI)
    float* __restrict__ y1, float* __restrict__ y1b)
{
  int idx = blockIdx.x * 256 + threadIdx.x;
  if (idx >= B_SZ * DI) return;
  int d = idx % DI;
  int b = idx / DI;
  float Af[NST], Ab[NST], h[NST], hb[NST];
  #pragma unroll
  for (int n = 0; n < NST; ++n) {
    Af[n] = -__expf(A_log[d * NST + n]);
    Ab[n] = -__expf(Ab_log[d * NST + n]);
    h[n] = 0.f; hb[n] = 0.f;
  }
  const float bd = b_dt[d];
  const size_t rowBase = (size_t)b * L_SZ;
  for (int l = 0; l < L_SZ; ++l) {
    size_t row  = rowBase + l;
    size_t rrow = rowBase + (L_SZ - 1 - l);
    float dr  = delta_raw[row * DI + d] + bd;
    float dlt = (dr > 20.f) ? dr : log1pf(__expf(dr));
    float uf  = u[row * DI + d];
    float ub  = u[rrow * DI + d];
    const float* bc = dbc + row * DBCW;
    float yf = 0.f, yb = 0.f;
    float du_f = dlt * uf, du_b = dlt * ub;
    #pragma unroll
    for (int n = 0; n < NST; ++n) {
      float Bn = bc[DTR + n];
      float Cn = bc[DTR + NST + n];
      h[n]  = __expf(dlt * Af[n]) * h[n]  + du_f * Bn;
      hb[n] = __expf(dlt * Ab[n]) * hb[n] + du_b * Bn;
      yf += h[n]  * Cn;
      yb += hb[n] * Cn;
    }
    y1[row * DI + d]   = yf;
    y1b[rrow * DI + d] = yb;  // reversed-back output
  }
}

// ---------------------------------------------------------------------------
// y = (y1 + y1b + t1) * silu(z1) -> bf16 for final GEMM
// t1 = h1[:, DI:2*DI], z1 = h1[:, 2*DI:3*DI]
// ---------------------------------------------------------------------------
__global__ __launch_bounds__(256) void k_gate_bf16(
    const float* __restrict__ y1, const float* __restrict__ y1b,
    const float* __restrict__ h1, __bf16* __restrict__ out)
{
  size_t idx = (size_t)blockIdx.x * 256 + threadIdx.x;
  const size_t total = (size_t)NTOK * DI;
  if (idx >= total) return;
  size_t r = idx / DI;
  int c = (int)(idx % DI);
  float t1 = h1[r * (size_t)(3 * DI) + DI + c];
  float z1 = h1[r * (size_t)(3 * DI) + 2 * DI + c];
  float s  = z1 / (1.f + __expf(-z1));
  out[idx] = (__bf16)((y1[idx] + y1b[idx] + t1) * s);
}

// ---------------------------------------------------------------------------
// host-side orchestration
// ---------------------------------------------------------------------------
extern "C" void kernel_launch(void* const* d_in, const int* in_sizes, int n_in,
                              void* d_out, int out_size, void* d_ws, size_t ws_size,
                              hipStream_t stream)
{
  const float* x1      = (const float*)d_in[0];
  const float* x2      = (const float*)d_in[1];
  const float* norm1_w = (const float*)d_in[2];
  const float* norm2_w = (const float*)d_in[3];
  const float* W_in1   = (const float*)d_in[4];
  const float* W_in2   = (const float*)d_in[5];
  const float* conv1_w = (const float*)d_in[6];
  const float* conv1_b = (const float*)d_in[7];
  const float* conv2_w = (const float*)d_in[8];
  const float* conv2_b = (const float*)d_in[9];
  const float* W_x2    = (const float*)d_in[10];
  const float* W_dt    = (const float*)d_in[11];
  const float* b_dt    = (const float*)d_in[12];
  const float* A2_log  = (const float*)d_in[13];
  const float* A2b_log = (const float*)d_in[14];
  const float* W_out   = (const float*)d_in[15];
  float* out = (float*)d_out;

  char* cur = (char*)d_ws;
  auto alloc = [&](size_t bytes) -> void* {
    void* r = (void*)cur;
    cur += (bytes + 255) & ~(size_t)255;
    return r;
  };

  __bf16* x1n  = (__bf16*)alloc((size_t)NTOK * DM * 2);
  __bf16* x2n  = (__bf16*)alloc((size_t)NTOK * DM * 2);
  __bf16* win1 = (__bf16*)alloc((size_t)3 * DI * DM * 2);
  __bf16* win2 = (__bf16*)alloc((size_t)DI * DM * 2);
  __bf16* wx2  = (__bf16*)alloc((size_t)DBCW * DI * 2);
  __bf16* wdt  = (__bf16*)alloc((size_t)DI * DTR * 2);
  __bf16* wout = (__bf16*)alloc((size_t)DM * DI * 2);
  float*  h1   = (float*) alloc((size_t)NTOK * 3 * DI * 4);
  float*  x2p  = (float*) alloc((size_t)NTOK * DI * 4);
  float*  uc   = (float*) alloc((size_t)NTOK * DI * 4);
  float*  x2c  = (float*) alloc((size_t)NTOK * DI * 4);
  __bf16* x2cb = (__bf16*)alloc((size_t)NTOK * DI * 2);
  float*  dbc  = (float*) alloc((size_t)NTOK * DBCW * 4);
  __bf16* drb  = (__bf16*)alloc((size_t)NTOK * DTR * 2);
  float*  draw = (float*) alloc((size_t)NTOK * DI * 4);
  float*  y1   = (float*) alloc((size_t)NTOK * DI * 4);
  float*  y1b  = (float*) alloc((size_t)NTOK * DI * 4);
  __bf16* ybf  = (__bf16*)alloc((size_t)NTOK * DI * 2);

  // 1) RMSNorm -> bf16
  k_rmsnorm_bf16<<<NTOK, 256, 0, stream>>>(x1, norm1_w, x1n, DM);
  k_rmsnorm_bf16<<<NTOK, 256, 0, stream>>>(x2, norm2_w, x2n, DM);

  // 2) weights -> bf16
  auto cvt = [&](const float* s, __bf16* dst, int rows, int cols) {
    size_t tot = (size_t)rows * cols;
    k_f32_to_bf16<<<(unsigned)((tot + 255) / 256), 256, 0, stream>>>(
        s, cols, dst, cols, rows, cols);
  };
  cvt(W_in1, win1, 3 * DI, DM);
  cvt(W_in2, win2, DI, DM);
  cvt(W_x2,  wx2,  DBCW, DI);
  cvt(W_dt,  wdt,  DI, DTR);
  cvt(W_out, wout, DM, DI);

  auto gemm = [&](const __bf16* Am, const __bf16* Wm, float* Cm, int M, int N, int K) {
    dim3 grid((unsigned)((N + TN - 1) / TN), (unsigned)((M + TM - 1) / TM));
    k_gemm_bf16_wmma<<<grid, 256, 0, stream>>>(Am, Wm, Cm, M, N, K);
  };

  // 3) projections
  gemm(x1n, win1, h1,  NTOK, 3 * DI, DM);   // h1 = [u | t1 | z1]
  gemm(x2n, win2, x2p, NTOK, DI, DM);

  // 4) depthwise causal convs (u is a strided slice of h1)
  size_t ntokdi = (size_t)NTOK * DI;
  unsigned gelem = (unsigned)((ntokdi + 255) / 256);
  k_dwconv_causal<<<gelem, 256, 0, stream>>>(h1, 3 * DI, conv1_w, conv1_b, uc, (__bf16*)nullptr);
  k_dwconv_causal<<<gelem, 256, 0, stream>>>(x2p, DI, conv2_w, conv2_b, x2c, x2cb);

  // 5) dbc = x2c @ W_x2^T ; delta_raw = d_r @ W_dt^T
  gemm(x2cb, wx2, dbc, NTOK, DBCW, DI);
  {
    size_t tot = (size_t)NTOK * DTR;
    k_f32_to_bf16<<<(unsigned)((tot + 255) / 256), 256, 0, stream>>>(
        dbc, DBCW, drb, DTR, NTOK, DTR);
  }
  gemm(drb, wdt, draw, NTOK, DI, DTR);

  // 6) bidirectional selective scan
  k_selective_scan_bidir<<<(B_SZ * DI + 255) / 256, 256, 0, stream>>>(
      uc, draw, dbc, A2_log, A2b_log, b_dt, y1, y1b);

  // 7) gate + output projection
  k_gate_bf16<<<gelem, 256, 0, stream>>>(y1, y1b, h1, ybf);
  gemm(ybf, wout, out, NTOK, DM, DI);

  (void)in_sizes; (void)n_in; (void)out_size; (void)ws_size;
}
